// Conv_MHSA_4569845203090
// MI455X (gfx1250) — matrix operand order repaired and checked by reading, not compiled, not run on hardware
//
#include <hip/hip_runtime.h>
#include <hip/hip_bf16.h>
#include <stdint.h>

typedef _Float16 v16h __attribute__((ext_vector_type(16)));
typedef _Float16 v8h  __attribute__((ext_vector_type(8)));
typedef float    v8f  __attribute__((ext_vector_type(8)));
typedef float    v4f  __attribute__((ext_vector_type(4)));

#define DEV __device__ __forceinline__

// ---- problem constants ----
static constexpr int   Dm   = 768;
static constexpr float QSCALE  = 0.125f * 1.44269504088896340736f; // hd^-0.5 * log2(e)
static constexpr float MASKADD = -1.442695040888963e12f;           // -1e12 * log2(e)

DEV v8f wmma16(v16h a, v16h b, v8f c) {
  return __builtin_amdgcn_wmma_f32_16x16x32_f16(false, a, false, b, (short)0, c, false, false);
}
DEV v16h mkfrag(const _Float16* p0, const _Float16* p1) {
  v8h lo = *(const v8h*)p0;
  v8h hi = *(const v8h*)p1;
  return __builtin_shufflevector(lo, hi, 0,1,2,3,4,5,6,7,8,9,10,11,12,13,14,15);
}
DEV v8f v8f_zero() {
  v8f z;
  #pragma unroll
  for (int i = 0; i < 8; ++i) z[i] = 0.f;
  return z;
}

template<int M>
DEV float shx(float v) { // xor-shuffle within 32 lanes via ds_swizzle (group-of-32 mode)
  return __builtin_bit_cast(float,
      __builtin_amdgcn_ds_swizzle(__builtin_bit_cast(int, v), (M << 10) | 0x1f));
}

struct Acc8 { v8f t[4][2]; };
DEV void acc_zero(Acc8& a) {
  #pragma unroll
  for (int i = 0; i < 4; ++i)
    #pragma unroll
    for (int j = 0; j < 2; ++j) a.t[i][j] = v8f_zero();
}

// 128x128 block tile GEMM mainloop: C[m][n] += A[m][k]*Bt[n][k], f16 in, f32 acc.
// 8 waves: wave (m2,n4), each wave 64x32 = 4x2 tiles of 16x16. LDS stride 40 halves
// (80B): 16B aligned ds_load_b128, conflict-free (gcd(20,64)-pattern distinct banks).
DEV void gemm128(const _Float16* __restrict__ A, int lda,
                 const _Float16* __restrict__ Bt, int ldb,
                 int m0, int n0, int K,
                 _Float16* As, _Float16* Bs, Acc8& acc)
{
  const int tid  = threadIdx.x;
  const int wave = tid >> 5, lane = tid & 31;
  const int wM = (wave & 1) * 64, wN = (wave >> 1) * 32;
  const int l15 = lane & 15, rh = lane >> 4;
  for (int ks = 0; ks < K; ks += 32) {
    __syncthreads();
    #pragma unroll
    for (int it = 0; it < 2; ++it) {
      int idx = tid + it * 256;
      int row = idx >> 2, ch = (idx & 3) * 8;
      *(v8h*)(As + row * 40 + ch) = *(const v8h*)(A + (size_t)(m0 + row) * lda + ks + ch);
      *(v8h*)(Bs + row * 40 + ch) = *(const v8h*)(Bt + (size_t)(n0 + row) * ldb + ks + ch);
    }
    __syncthreads();
    v16h af[4], bf[2];
    #pragma unroll
    for (int mi = 0; mi < 4; ++mi) {      // A frag: row per lane, K halves {kOff..+7,+16..+23}
      const _Float16* p = As + (wM + mi * 16 + l15) * 40 + rh * 8;
      af[mi] = mkfrag(p, p + 16);
    }
    #pragma unroll
    for (int ni = 0; ni < 2; ++ni) {      // B frag: col per lane, K halves {kOff..+7,+16..+23}
      const _Float16* p = Bs + (wN + ni * 16 + l15) * 40 + rh * 8;
      bf[ni] = mkfrag(p, p + 16);
    }
    #pragma unroll
    for (int mi = 0; mi < 4; ++mi)
      #pragma unroll
      for (int ni = 0; ni < 2; ++ni)
        acc.t[mi][ni] = wmma16(af[mi], bf[ni], acc.t[mi][ni]);
  }
}

// ---------------- prep kernels ----------------
__global__ void cvt_f16(const float* __restrict__ s, _Float16* __restrict__ d, int n) {
  int i = blockIdx.x * 256 + threadIdx.x;
  if (i < n) d[i] = (_Float16)s[i];
}
__global__ void wt_transpose(const float* __restrict__ W, _Float16* __restrict__ Wt) {
  int i = blockIdx.x * 256 + threadIdx.x;          // 2304*768 = 1769472
  int nIdx = i / 768, kk = i - nIdx * 768;
  Wt[i] = (_Float16)W[(size_t)kk * 2304 + nIdx];
}

// ---------------- QKV GEMM: writes q (scaled), k [b,h,n,64] and vT [b,h,64,n] ----------------
__global__ __launch_bounds__(256) void qkv_gemm(const _Float16* __restrict__ xh,
                                                const _Float16* __restrict__ Wt,
                                                _Float16* __restrict__ q,
                                                _Float16* __restrict__ k,
                                                _Float16* __restrict__ vT)
{
  __shared__ __align__(16) _Float16 As[128 * 40];
  __shared__ __align__(16) _Float16 Bs[128 * 40];
  const int m0 = blockIdx.x * 128, n0 = blockIdx.y * 128;
  Acc8 acc; acc_zero(acc);
  gemm128(xh, Dm, Wt, Dm, m0, n0, Dm, As, Bs, acc);

  const int wave = threadIdx.x >> 5, lane = threadIdx.x & 31;
  const int wM = (wave & 1) * 64, wN = (wave >> 1) * 32;
  const int l15 = lane & 15, rh = lane >> 4;
  #pragma unroll
  for (int mi = 0; mi < 4; ++mi) {
    const int rowBase = m0 + wM + mi * 16 + rh * 8;   // never crosses a batch (128|1024)
    const int batch = rowBase >> 10, tok = rowBase & 1023;
    #pragma unroll
    for (int ni = 0; ni < 2; ++ni) {
      const int gcol = n0 + wN + ni * 16 + l15;       // 0..2303
      const int sect = gcol / 768;                    // uniform per tile (16 | 768)
      const int within = gcol - sect * 768;
      const int bh = batch * 12 + (within >> 6);
      const int hd = within & 63;
      v8f a = acc.t[mi][ni];
      if (sect == 0) {
        _Float16* p = q + (((size_t)bh << 10) + tok) * 64 + hd;
        #pragma unroll
        for (int r = 0; r < 8; ++r) p[r * 64] = (_Float16)(a[r] * QSCALE);
      } else if (sect == 1) {
        _Float16* p = k + (((size_t)bh << 10) + tok) * 64 + hd;
        #pragma unroll
        for (int r = 0; r < 8; ++r) p[r * 64] = (_Float16)a[r];
      } else {
        _Float16* p = vT + (((size_t)bh << 6) + hd) * 1024 + tok;
        v8h pk;
        #pragma unroll
        for (int r = 0; r < 8; ++r) pk[r] = (_Float16)a[r];
        *(v8h*)p = pk;                                 // 8 consecutive tokens, 16B store
      }
    }
  }
}

// ---------------- flash attention ----------------
__global__ __launch_bounds__(256) void attn_kernel(const _Float16* __restrict__ q,
                                                   const _Float16* __restrict__ k,
                                                   const _Float16* __restrict__ vT,
                                                   const float* __restrict__ mask,
                                                   float* __restrict__ out)
{
  __shared__ __align__(16) _Float16 pbuf[8 * 16 * 72];  // per-wave P tile, padded stride
  const int bh = blockIdx.x >> 3;                       // 0..47 = b*12+h
  const int qt = blockIdx.x & 7;
  const int wave = threadIdx.x >> 5, lane = threadIdx.x & 31;
  const int l15 = lane & 15, rh = lane >> 4;
  const int qRow = qt * 128 + wave * 16;

  const _Float16* qp = q + (((size_t)bh << 10) + qRow + l15) * 64;
  v16h qa0 = mkfrag(qp + rh * 16,      qp + rh * 16 + 8);
  v16h qa1 = mkfrag(qp + 32 + rh * 16, qp + 32 + rh * 16 + 8);

  float m[8], l[8];
  v8f o[4];
  #pragma unroll
  for (int r = 0; r < 8; ++r) { m[r] = -3.0e38f; l[r] = 0.f; }
  #pragma unroll
  for (int dt = 0; dt < 4; ++dt) o[dt] = v8f_zero();

  _Float16* pb = pbuf + wave * (16 * 72);
  const float*    mrow = mask + ((size_t)bh << 20) + ((size_t)(qRow + rh * 8) << 10) + l15;
  const _Float16* kbh  = k  + (((size_t)bh << 10) + l15) * 64 + rh * 16;
  const _Float16* vbh  = vT + ((size_t)bh << 16) + ((size_t)l15 << 10) + rh * 16;

  for (int jc = 0; jc < 1024; jc += 64) {               // 64-key chunks
    v8f s[4];
    #pragma unroll
    for (int t = 0; t < 4; ++t) {                       // S = q . k^T  (2 WMMA per j-tile)
      const int jt = jc + t * 16;
      const _Float16* kp = kbh + (size_t)jt * 64;
      v16h kb0 = mkfrag(kp,      kp + 8);
      v16h kb1 = mkfrag(kp + 32, kp + 40);
      v8f c = wmma16(qa1, kb1, v8f_zero());
      s[t] = wmma16(qa0, kb0, c);
      const float* mp = mrow + jt;
      #pragma unroll
      for (int r = 0; r < 8; ++r)                       // dropkey additive mask
        s[t][r] += (mp[(size_t)r << 10] < 0.1f) ? MASKADD : 0.0f;
    }
    float tm[8], corr[8], rs[8];
    #pragma unroll
    for (int r = 0; r < 8; ++r)
      tm[r] = fmaxf(fmaxf(s[0][r], s[1][r]), fmaxf(s[2][r], s[3][r]));
    #pragma unroll
    for (int r = 0; r < 8; ++r) {                       // row-max across 16 lanes
      tm[r] = fmaxf(tm[r], shx<1>(tm[r]));
      tm[r] = fmaxf(tm[r], shx<2>(tm[r]));
      tm[r] = fmaxf(tm[r], shx<4>(tm[r]));
      tm[r] = fmaxf(tm[r], shx<8>(tm[r]));
    }
    #pragma unroll
    for (int r = 0; r < 8; ++r) {
      float mn = fmaxf(m[r], tm[r]);
      corr[r] = __builtin_amdgcn_exp2f(m[r] - mn);
      m[r] = mn;
      float sum = 0.f;
      #pragma unroll
      for (int t = 0; t < 4; ++t) {
        float pv = __builtin_amdgcn_exp2f(s[t][r] - mn);
        s[t][r] = pv;
        sum += pv;
      }
      rs[r] = sum;
    }
    #pragma unroll
    for (int r = 0; r < 8; ++r) {                       // row-sum across 16 lanes
      rs[r] += shx<1>(rs[r]);
      rs[r] += shx<2>(rs[r]);
      rs[r] += shx<4>(rs[r]);
      rs[r] += shx<8>(rs[r]);
      l[r] = l[r] * corr[r] + rs[r];
    }
    #pragma unroll
    for (int dt = 0; dt < 4; ++dt)
      #pragma unroll
      for (int r = 0; r < 8; ++r) o[dt][r] *= corr[r];

    // D-layout -> A-layout via per-wave LDS (DS ops are in-order per wave)
    #pragma unroll
    for (int t = 0; t < 4; ++t)
      #pragma unroll
      for (int r = 0; r < 8; ++r)
        pb[(rh * 8 + r) * 72 + t * 16 + l15] = (_Float16)s[t][r];

    const _Float16* pp = pb + l15 * 72 + rh * 16;
    v16h pa0 = mkfrag(pp,      pp + 8);
    v16h pa1 = mkfrag(pp + 32, pp + 40);

    #pragma unroll
    for (int dt = 0; dt < 4; ++dt) {                    // O += P . V  (2 WMMA per d-tile)
      const _Float16* vp = vbh + ((size_t)dt << 14) + jc;
      v16h vb0 = mkfrag(vp,      vp + 8);
      v16h vb1 = mkfrag(vp + 32, vp + 40);
      v8f t2 = wmma16(pa1, vb1, o[dt]);
      o[dt] = wmma16(pa0, vb0, t2);
    }
  }
  const int batch = bh / 12, head = bh - batch * 12;
  #pragma unroll
  for (int r = 0; r < 8; ++r) {
    float inv = __builtin_amdgcn_rcpf(l[r]);
    float* op = out + ((size_t)(batch * 1024 + qRow + rh * 8 + r)) * 768 + head * 64 + l15;
    #pragma unroll
    for (int dt = 0; dt < 4; ++dt) op[dt * 16] = o[dt][r] * inv;
  }
}

// ---------------- conv branch ----------------
__global__ void dwconv(const float* __restrict__ y, const float* __restrict__ w1,
                       _Float16* __restrict__ y1h)
{
  int tid = blockIdx.x * 256 + threadIdx.x;             // 4*256*1024
  int p  = tid & 1023;
  int ic = (tid >> 10) & 255;
  int b  = tid >> 18;
  int hh = p >> 5, ww = p & 31;
  const float* yp = y + ((size_t)(b * 256 + ic) << 10);
  const float* wp = w1 + ic * 9;
  float sum = 0.f;
  #pragma unroll
  for (int dy = -1; dy <= 1; ++dy) {
    int hy = hh + dy;
    if ((unsigned)hy < 32u) {
      #pragma unroll
      for (int dx = -1; dx <= 1; ++dx) {
        int wx = ww + dx;
        if ((unsigned)wx < 32u) sum += wp[(dy + 1) * 3 + (dx + 1)] * yp[hy * 32 + wx];
      }
    }
  }
  y1h[((size_t)(b * 1024 + p)) * 256 + ic] = (_Float16)sum;  // NHWC f16 for GEMM A
}

__global__ __launch_bounds__(256) void conv1x1_gemm(const _Float16* __restrict__ y1h,
                                                    const _Float16* __restrict__ w2h,
                                                    float* __restrict__ y2t)
{
  __shared__ __align__(16) _Float16 As[128 * 40];
  __shared__ __align__(16) _Float16 Bs[128 * 40];
  const int m0 = blockIdx.x * 128, n0 = blockIdx.y * 128;
  Acc8 acc; acc_zero(acc);
  gemm128(y1h, 256, w2h, 256, m0, n0, 256, As, Bs, acc);

  const int wave = threadIdx.x >> 5, lane = threadIdx.x & 31;
  const int wM = (wave & 1) * 64, wN = (wave >> 1) * 32;
  const int l15 = lane & 15, rh = lane >> 4;
  #pragma unroll
  for (int mi = 0; mi < 4; ++mi) {
    const int rowBase = m0 + wM + mi * 16 + rh * 8;
    #pragma unroll
    for (int ni = 0; ni < 2; ++ni) {
      const int oc = n0 + wN + ni * 16 + l15;
      float* p = y2t + (size_t)oc * 4096 + rowBase;     // channel-major for BN
      v8f a = acc.t[mi][ni];
      *(v4f*)p       = __builtin_shufflevector(a, a, 0, 1, 2, 3);
      *(v4f*)(p + 4) = __builtin_shufflevector(a, a, 4, 5, 6, 7);
    }
  }
}

__global__ void bn_stats(const float* __restrict__ y2t, const float* __restrict__ g,
                         const float* __restrict__ be, float* __restrict__ scl,
                         float* __restrict__ bia)
{
  __shared__ float ss[8], sq[8];
  const int c = blockIdx.x, t = threadIdx.x;
  const float* p = y2t + (size_t)c * 4096;
  float s = 0.f, q2 = 0.f;
  for (int i = t; i < 4096; i += 256) { float v = p[i]; s += v; q2 += v * v; }
  s  += shx<1>(s);  s  += shx<2>(s);  s  += shx<4>(s);  s  += shx<8>(s);  s  += shx<16>(s);
  q2 += shx<1>(q2); q2 += shx<2>(q2); q2 += shx<4>(q2); q2 += shx<8>(q2); q2 += shx<16>(q2);
  if ((t & 31) == 0) { ss[t >> 5] = s; sq[t >> 5] = q2; }
  __syncthreads();
  if (t == 0) {
    float S = 0.f, Q = 0.f;
    #pragma unroll
    for (int w = 0; w < 8; ++w) { S += ss[w]; Q += sq[w]; }
    float mean = S * (1.f / 4096.f);
    float var  = Q * (1.f / 4096.f) - mean * mean;
    float sc   = g[c] * __builtin_amdgcn_rsqf(var + 1e-5f);
    scl[c] = sc;
    bia[c] = be[c] - mean * sc;
  }
}

__global__ void final_yout(const float* __restrict__ y2t, const float* __restrict__ scl,
                           const float* __restrict__ bia, float* __restrict__ yo)
{
  int tid = blockIdx.x * 256 + threadIdx.x;             // 768*4096, (c, m) order
  int c = tid >> 12;
  int mm = tid & 4095;
  float v = fmaxf(y2t[tid] * scl[c] + bia[c], 0.f);
  yo[(((size_t)(mm >> 10) * 768 + c) << 10) + (mm & 1023)] = v;   // NCHW y_out
}

__global__ void final_add(const float* __restrict__ y2t, const float* __restrict__ scl,
                          const float* __restrict__ bia, float* __restrict__ out)
{
  int tid = blockIdx.x * 256 + threadIdx.x;             // 4096*768, (m, c) order
  int mm = tid / 768;
  int c = tid - mm * 768;
  float v = fmaxf(y2t[(size_t)c * 4096 + mm] * scl[c] + bia[c], 0.f);
  out[tid] += v;                                        // attn out written earlier this call
}

// ---------------- launcher ----------------
extern "C" void kernel_launch(void* const* d_in, const int* in_sizes, int n_in,
                              void* d_out, int out_size, void* d_ws, size_t ws_size,
                              hipStream_t stream)
{
  (void)in_sizes; (void)n_in; (void)out_size; (void)ws_size;
  const float* x     = (const float*)d_in[0];
  const float* y     = (const float*)d_in[1];
  const float* dmask = (const float*)d_in[2];
  const float* Wq    = (const float*)d_in[3];
  const float* w1    = (const float*)d_in[4];
  const float* w2    = (const float*)d_in[5];
  const float* gamma = (const float*)d_in[6];
  const float* beta  = (const float*)d_in[7];
  float* out = (float*)d_out;

  char* ws = (char*)d_ws;                                // ~41.8 MiB total
  _Float16* xh  = (_Float16*)(ws);                       // 4096*768   f16
  _Float16* Wt  = (_Float16*)(ws + 6291456);             // 2304*768   f16
  _Float16* qws = (_Float16*)(ws + 9830400);             // 48*1024*64 f16
  _Float16* kws = (_Float16*)(ws + 16121856);
  _Float16* vT  = (_Float16*)(ws + 22413312);
  _Float16* y1h = (_Float16*)(ws + 28704768);            // 4096*256   f16
  _Float16* w2h = (_Float16*)(ws + 30801920);            // 768*256    f16
  float*    y2t = (float*)(ws + 31195136);               // 768*4096   f32
  float*    scl = (float*)(ws + 43778048);
  float*    bia = (float*)(ws + 43781120);

  cvt_f16<<<12288, 256, 0, stream>>>(x, xh, 3145728);
  wt_transpose<<<6912, 256, 0, stream>>>(Wq, Wt);
  cvt_f16<<<768, 256, 0, stream>>>(w2, w2h, 196608);

  qkv_gemm<<<dim3(32, 18), 256, 0, stream>>>(xh, Wt, qws, kws, vT);
  attn_kernel<<<384, 256, 0, stream>>>(qws, kws, vT, dmask, out);

  dwconv<<<4096, 256, 0, stream>>>(y, w1, y1h);
  conv1x1_gemm<<<dim3(32, 6), 256, 0, stream>>>(y1h, w2h, y2t);
  bn_stats<<<768, 256, 0, stream>>>(y2t, gamma, beta, scl, bia);
  final_yout<<<12288, 256, 0, stream>>>(y2t, scl, bia, out + 3145728);
  final_add<<<12288, 256, 0, stream>>>(y2t, scl, bia, out);
}
